// GroupedQueryAttention_30210799960658
// MI455X (gfx1250) — matrix operand order repaired and checked by reading, not compiled, not run on hardware
//
#include <hip/hip_runtime.h>

// ---------------------------------------------------------------------------
// Grouped-Query Attention, CDNA5 (gfx1250), bf16 WMMA pipeline.
//   B=4, T=2048, C=1024, HQ=16, HKV=4, D=64, G=4
// Outputs (concatenated in d_out): y [B,T,C] fp32, att [B,G,HKV,T,T] fp32.
//
// v2: V projection now writes V^T ([b,h,d,t], t-contiguous) so the P@V
//     B-fragments are contiguous 16B vector loads instead of 16 strided
//     2-byte loads; phase 3 uses all 8 waves (even/odd K strips + LDS
//     partial reduction).
// ---------------------------------------------------------------------------

typedef __bf16 bf16;
typedef __bf16 v16bf __attribute__((ext_vector_type(16)));
typedef __bf16 v8bf  __attribute__((ext_vector_type(8)));
typedef float  v8f   __attribute__((ext_vector_type(8)));

constexpr int Bn  = 4;
constexpr int Tn  = 2048;
constexpr int Cn  = 1024;
constexpr int HQn = 16;
constexpr int HKVn = 4;
constexpr int Dn  = 64;
constexpr int Gn  = 4;
constexpr int KVn = HKVn * Dn;   // 256
constexpr int SROW = Tn + 8;     // padded LDS row stride (floats)

#define WMMA_BF16(a, b, c) \
  __builtin_amdgcn_wmma_f32_16x16x32_bf16(false, (a), false, (b), (short)0, (c), false, false)

// --- WMMA fragment loaders (wave32 layouts per CDNA5 ISA 7.12.2) ------------
// A (16x32 bf16): lanes 0-15 row M=lane hold K {0..7,16..23}; lanes 16-31 hold
// K {8..15,24..31}.  A row-major => two contiguous 16B loads per lane.
__device__ __forceinline__ v16bf load_frag_a(const bf16* __restrict__ A, int lda,
                                             int row, int k0, int lane) {
  const bf16* p = A + (size_t)row * lda + k0 + ((lane & 16) ? 8 : 0);
  v8bf lo = *(const v8bf*)p;
  v8bf hi = *(const v8bf*)(p + 16);
  v16bf r;
#pragma unroll
  for (int i = 0; i < 8; ++i) { r[i] = lo[i]; r[i + 8] = hi[i]; }
  return r;
}

// B (32x16 bf16): lanes 0-15 col N=lane hold K {0..7,16..23}; lanes 16-31 hold
// K {8..15,24..31} (matches A).  Column-contiguous => two 16B loads per lane.
__device__ __forceinline__ v16bf load_frag_b(const bf16* __restrict__ colptr,
                                             int k0, int lane) {
  const bf16* p = colptr + k0 + ((lane & 16) ? 8 : 0);
  v8bf lo = *(const v8bf*)p;
  v8bf hi = *(const v8bf*)(p + 16);
  v16bf r;
#pragma unroll
  for (int i = 0; i < 8; ++i) { r[i] = lo[i]; r[i + 8] = hi[i]; }
  return r;
}

// --- elementwise conversions ------------------------------------------------
__global__ void f32_to_bf16(const float* __restrict__ in, bf16* __restrict__ out,
                            size_t n) {
  size_t i = (size_t)blockIdx.x * blockDim.x + threadIdx.x;
  if (i < n) out[i] = (bf16)in[i];
}

// in: [K,N] row-major -> out: [N,K] (column-contiguous for WMMA B fragments)
__global__ void transpose_f32_to_bf16(const float* __restrict__ in,
                                      bf16* __restrict__ out, int K, int N) {
  size_t i = (size_t)blockIdx.x * blockDim.x + threadIdx.x;
  if (i < (size_t)K * N) {
    int n = (int)(i / K), k = (int)(i % K);
    out[i] = (bf16)in[(size_t)k * N + n];
  }
}

// --- generic bf16 WMMA GEMM:  out = (A @ Wt^T + bias) * scale ---------------
// A: [M,K] bf16 row-major.  Wt: [N,K] bf16 (pre-transposed weight).
enum { MODE_Q = 0, MODE_KV = 1, MODE_VT = 2, MODE_F32 = 3 };

template <int MODE>
__global__ __launch_bounds__(256) void gemm_bf16_wmma(
    const bf16* __restrict__ A, const bf16* __restrict__ Wt,
    const float* __restrict__ bias, void* __restrict__ out, int M, int N, int K,
    float scale) {
  const int lane = threadIdx.x & 31;
  const int wave = threadIdx.x >> 5;
  const int tilesN = N >> 4;
  const int tile = blockIdx.x * 8 + wave;
  if (tile >= (M >> 4) * tilesN) return;  // wave-uniform
  const int tm = tile / tilesN, tn = tile % tilesN;

  const int arow = tm * 16 + (lane & 15);
  const bf16* colptr = Wt + (size_t)(tn * 16 + (lane & 15)) * K;

  v8f acc = {};
  for (int k0 = 0; k0 < K; k0 += 32) {
    v16bf a = load_frag_a(A, K, arow, k0, lane);
    v16bf b = load_frag_b(colptr, k0, lane);
    acc = WMMA_BF16(a, b, acc);
  }

  const int col = tn * 16 + (lane & 15);
  const float bcol = bias ? bias[col] : 0.0f;
#pragma unroll
  for (int r = 0; r < 8; ++r) {
    const int row = tm * 16 + r + ((lane & 16) ? 8 : 0);
    float v = (acc[r] + bcol) * scale;
    if (MODE == MODE_F32) {
      ((float*)out)[(size_t)row * N + col] = v;
    } else {
      const int bi = row >> 11, t = row & (Tn - 1);
      const int hd = col >> 6, d = col & (Dn - 1);
      if (MODE == MODE_VT) {
        // V^T: [b, h, d, t] (t contiguous) for vectorized P@V B-fragments
        ((bf16*)out)[((size_t)(bi * HKVn + hd) * Dn + d) * Tn + t] = (bf16)v;
      } else {
        const int H = (MODE == MODE_Q) ? HQn : HKVn;
        ((bf16*)out)[(((size_t)(bi * H + hd) * Tn) + t) * Dn + d] = (bf16)v;
      }
    }
  }
}

// --- fused attention: scores -> softmax -> att store -> P@V -> ctx ----------
// One workgroup (8 waves) per (b, hq, 16-row query block).  The 16xT fp32
// score strip lives in (dynamic) LDS; att is written exactly once.
__global__ __launch_bounds__(256) void gqa_fused_attn(
    const bf16* __restrict__ qb, const bf16* __restrict__ kb,
    const bf16* __restrict__ vt, float* __restrict__ att,
    bf16* __restrict__ ctx) {
  extern __shared__ float smem[];
  float* S    = smem;                 // 16 * SROW scores / probs
  float* rred = S + 16 * SROW;        // 16 * 17 partial reductions
  float* rowM = rred + 16 * 17;       // 16 row maxima
  float* rowS = rowM + 16;            // 16 reciprocal row sums
  float* PS   = rowS + 16;            // 4 * 256 P@V partial accumulators

  const int tid = threadIdx.x;
  const int lane = tid & 31, wave = tid >> 5;
  const int q0 = blockIdx.x;          // query tile (16 rows)
  const int hq = blockIdx.y;          // query head = h*G + g
  const int b  = blockIdx.z;
  const int h = hq >> 2, g = hq & 3;  // G = 4

  const bf16* qbase  = qb + ((size_t)(b * HQn + hq)) * Tn * Dn;
  const bf16* kbase  = kb + ((size_t)(b * HKVn + h)) * Tn * Dn;
  const bf16* vtbase = vt + ((size_t)(b * HKVn + h)) * Dn * Tn;

  const int kmax = (q0 + 1) * 16;          // causal extent (multiple of 16)
  const int kpad = (kmax + 31) & ~31;      // rounded to WMMA K-strip

  // phase 1: S = Q K^T (1/sqrt(D) folded into Q), causal mask on diagonal tile
  const int qrow = q0 * 16 + (lane & 15);
  const v16bf aq0 = load_frag_a(qbase, Dn, qrow, 0, lane);
  const v16bf aq1 = load_frag_a(qbase, Dn, qrow, 32, lane);

  for (int kt = wave; kt <= q0; kt += 8) {
    const bf16* colK = kbase + (size_t)(kt * 16 + (lane & 15)) * Dn;
    v8f acc = {};
    acc = WMMA_BF16(aq0, load_frag_b(colK, 0, lane), acc);
    acc = WMMA_BF16(aq1, load_frag_b(colK, 32, lane), acc);
    const int kc = kt * 16 + (lane & 15);
#pragma unroll
    for (int r = 0; r < 8; ++r) {
      const int rr = r + ((lane & 16) ? 8 : 0);
      S[rr * SROW + kc] = (kc <= q0 * 16 + rr) ? acc[r] : -1e30f;
    }
  }
  if (kpad > kmax) {  // zero the 16-col pad so the last P@V strip adds nothing
    S[(tid >> 4) * SROW + kmax + (tid & 15)] = 0.0f;
  }
  __syncthreads();

  // phase 2: row softmax over [0,kmax).  row = tid>>4, 16 threads per row.
  const int row = tid >> 4, chunk = tid & 15;
  float* Srow = S + row * SROW;
  float pm = -1e30f;
  for (int c = chunk; c < kmax; c += 16) pm = fmaxf(pm, Srow[c]);
  rred[row * 17 + chunk] = pm;
  __syncthreads();
  if (tid < 16) {
    float m = -1e30f;
    for (int i = 0; i < 16; ++i) m = fmaxf(m, rred[tid * 17 + i]);
    rowM[tid] = m;
  }
  __syncthreads();
  const float m = rowM[row];
  float ps = 0.0f;
  for (int c = chunk; c < kmax; c += 16) {
    float e = __expf(Srow[c] - m);
    Srow[c] = e;
    ps += e;
  }
  rred[row * 17 + chunk] = ps;
  __syncthreads();
  if (tid < 16) {
    float s = 0.0f;
    for (int i = 0; i < 16; ++i) s += rred[tid * 17 + i];
    rowS[tid] = 1.0f / s;
  }
  __syncthreads();
  // normalize in LDS and emit att[b,g,h,q,:] exactly once (coalesced per row)
  const float inv = rowS[row];
  float* attRow =
      att + (((size_t)((b * Gn + g) * HKVn + h) * Tn + (q0 * 16 + row)) * Tn);
  for (int c = chunk; c < kmax; c += 16) {
    float p = Srow[c] * inv;
    Srow[c] = p;
    attRow[c] = p;
  }
  for (int c = kmax + chunk; c < Tn; c += 16) attRow[c] = 0.0f;
  __syncthreads();

  // phase 3: ctx(16x64) = P(16xkpad) @ V(kpad x 64).
  // Wave w owns d-tile (w&3); waves 0-3 take even 32-K strips, 4-7 odd.
  const int ntile = wave & 3;
  const int dcol = ntile * 16 + (lane & 15);
  const bf16* colV = vtbase + (size_t)dcol * Tn;  // V^T column, k-contiguous
  const int abase = (lane & 16) ? 8 : 0;
  const int hi8 = (lane & 16) ? 8 : 0;
  v8f acc = {};
  for (int k32 = (wave >> 2) * 32; k32 < kpad; k32 += 64) {
    v16bf pa;
#pragma unroll
    for (int i = 0; i < 16; ++i) {
      const int kk = k32 + abase + (i < 8 ? i : i + 8);
      pa[i] = (bf16)S[(lane & 15) * SROW + kk];
    }
    v16bf vf = load_frag_b(colV, k32, lane);
    acc = WMMA_BF16(pa, vf, acc);
  }
  __syncthreads();
  if (wave >= 4) {  // publish odd-strip partials
#pragma unroll
    for (int r = 0; r < 8; ++r)
      PS[ntile * 256 + (r + hi8) * 16 + (lane & 15)] = acc[r];
  }
  __syncthreads();
  if (wave < 4) {   // combine and store ctx (bf16, [b, t, hq*D + d])
#pragma unroll
    for (int r = 0; r < 8; ++r) {
      const float v = acc[r] + PS[ntile * 256 + (r + hi8) * 16 + (lane & 15)];
      const int t = q0 * 16 + r + hi8;
      ctx[((size_t)(b * Tn + t)) * Cn + hq * Dn + dcol] = (bf16)v;
    }
  }
}

// ---------------------------------------------------------------------------
extern "C" void kernel_launch(void* const* d_in, const int* in_sizes, int n_in,
                              void* d_out, int out_size, void* d_ws,
                              size_t ws_size, hipStream_t stream) {
  (void)in_sizes; (void)n_in; (void)out_size; (void)ws_size;
  const float* x  = (const float*)d_in[0];
  const float* Wq = (const float*)d_in[1];
  const float* bq = (const float*)d_in[2];
  const float* Wk = (const float*)d_in[3];
  const float* bk = (const float*)d_in[4];
  const float* Wv = (const float*)d_in[5];
  const float* bv = (const float*)d_in[6];
  const float* Wo = (const float*)d_in[7];
  const float* bo = (const float*)d_in[8];

  float* y_out   = (float*)d_out;                  // [B,T,C]
  float* att_out = y_out + (size_t)Bn * Tn * Cn;   // [B,G,HKV,T,T]

  char* w = (char*)d_ws;
  bf16* xb  = (bf16*)w; w += (size_t)Bn * Tn * Cn * 2;
  bf16* WqT = (bf16*)w; w += (size_t)Cn * Cn * 2;
  bf16* WkT = (bf16*)w; w += (size_t)Cn * KVn * 2;
  bf16* WvT = (bf16*)w; w += (size_t)Cn * KVn * 2;
  bf16* WoT = (bf16*)w; w += (size_t)Cn * Cn * 2;
  bf16* qb  = (bf16*)w; w += (size_t)Bn * Tn * Cn * 2;
  bf16* kb  = (bf16*)w; w += (size_t)Bn * HKVn * Tn * Dn * 2;
  bf16* vt  = (bf16*)w; w += (size_t)Bn * HKVn * Tn * Dn * 2;
  bf16* ctx = (bf16*)w; w += (size_t)Bn * Tn * Cn * 2;   // ~64 MB total

  const size_t nx = (size_t)Bn * Tn * Cn;
  f32_to_bf16<<<(unsigned)((nx + 255) / 256), 256, 0, stream>>>(x, xb, nx);
  transpose_f32_to_bf16<<<(Cn * Cn + 255) / 256, 256, 0, stream>>>(Wq, WqT, Cn, Cn);
  transpose_f32_to_bf16<<<(Cn * KVn + 255) / 256, 256, 0, stream>>>(Wk, WkT, Cn, KVn);
  transpose_f32_to_bf16<<<(Cn * KVn + 255) / 256, 256, 0, stream>>>(Wv, WvT, Cn, KVn);
  transpose_f32_to_bf16<<<(Cn * Cn + 255) / 256, 256, 0, stream>>>(Wo, WoT, Cn, Cn);

  const int M = Bn * Tn;  // 8192
  gemm_bf16_wmma<MODE_Q><<<(M / 16) * (Cn / 16) / 8, 256, 0, stream>>>(
      xb, WqT, bq, qb, M, Cn, Cn, 0.125f);  // 1/sqrt(D) folded in
  gemm_bf16_wmma<MODE_KV><<<(M / 16) * (KVn / 16) / 8, 256, 0, stream>>>(
      xb, WkT, bk, kb, M, KVn, Cn, 1.0f);
  gemm_bf16_wmma<MODE_VT><<<(M / 16) * (KVn / 16) / 8, 256, 0, stream>>>(
      xb, WvT, bv, vt, M, KVn, Cn, 1.0f);

  const size_t shmem =
      (size_t)(16 * SROW + 16 * 17 + 32 + 4 * 256) * sizeof(float);
  gqa_fused_attn<<<dim3(Tn / 16, HQn, Bn), 256, shmem, stream>>>(qb, kb, vt,
                                                                 att_out, ctx);

  gemm_bf16_wmma<MODE_F32><<<(M / 16) * (Cn / 16) / 8, 256, 0, stream>>>(
      ctx, WoT, bo, y_out, M, Cn, Cn, 1.0f);
}
